// GCNEncoder_38500086841928
// MI455X (gfx1250) — matrix operand order repaired and checked by reading, not compiled, not run on hardware
//
#include <hip/hip_runtime.h>

#define IN_DIM 128
#define HID    64
#define EMB    128

// LDS column strides (bf16 elements), padded +8 (16B) so that the per-lane
// fragment reads land on distinct bank groups (stride 68/36 dwords -> lane
// bank offset 4*lane mod 64, each b128 covering its own 4 banks).
#define S1 (IN_DIM + 8)   // gemm1: 136 bf16 = 272 B per column
#define S2 (HID + 8)      // gemm2:  72 bf16 = 144 B per column

typedef __attribute__((ext_vector_type(16))) __bf16 v16bf;
typedef __attribute__((ext_vector_type(8)))  __bf16 v8bf;
typedef __attribute__((ext_vector_type(8)))  float  v8f;

__device__ __forceinline__ __bf16 to_bf16(float f) { return (__bf16)f; }

__device__ __forceinline__ v8f wmma_bf16(v16bf a, v16bf b, v8f c) {
    // D = A(16x32 bf16) x B(32x16 bf16) + C(16x16 f32)
    return __builtin_amdgcn_wmma_f32_16x16x32_bf16(
        false, a, false, b, (short)0, c, false, false);
}

// Load 16 contiguous bf16 (32 B, 16B-aligned) as two ds_load_b128.
__device__ __forceinline__ v16bf load_bfrag(const __bf16* p) {
    v8bf lo = *(const v8bf*)p;
    v8bf hi = *(const v8bf*)(p + 8);
    return __builtin_shufflevector(lo, hi, 0, 1, 2, 3, 4, 5, 6, 7,
                                           8, 9, 10, 11, 12, 13, 14, 15);
}

// ---------------------------------------------------------------- norm ----
__global__ __launch_bounds__(256) void init_deg_kernel(float* deg, int n) {
    int i = blockIdx.x * 256 + threadIdx.x;
    if (i < n) deg[i] = 1.0f;   // self-loop weight
}

__global__ __launch_bounds__(256) void deg_accum_kernel(const int* __restrict__ dst,
                                                        const float* __restrict__ w,
                                                        float* __restrict__ deg, int E) {
    int e = blockIdx.x * 256 + threadIdx.x;
    if (e < E) atomicAdd(&deg[dst[e]], w[e]);
}

__global__ __launch_bounds__(256) void dinv_kernel(float* deg, int n) {
    int i = blockIdx.x * 256 + threadIdx.x;
    if (i < n) deg[i] = rsqrtf(deg[i]);   // deg >= 1 always (self-loop)
}

// --------------------------------------------------------------- GEMM 1 ---
// h1 = x @ W1 ; agg1 = h1 * dinv^2 (self-loop seed of the segment sum)
__global__ __launch_bounds__(256) void gemm1_kernel(
    const float* __restrict__ x, const float* __restrict__ W1,
    const float* __restrict__ dinv,
    float* __restrict__ h1, float* __restrict__ agg1, int n)
{
    __shared__ __bf16 wlds[HID * S1];   // column-major W1: wlds[col*S1 + k]
    for (int i = threadIdx.x; i < IN_DIM * HID; i += 256) {
        int k = i >> 6, col = i & 63;           // W1 is [k][col], row-major
        wlds[col * S1 + k] = to_bf16(W1[i]);
    }
    __syncthreads();

    const int wave = threadIdx.x >> 5;
    const int lane = threadIdx.x & 31;
    const int half = lane >> 4;
    const int nloc = lane & 15;
    const int rowBase = blockIdx.x * 128 + wave * 16;

    int arow = rowBase + (lane & 15);
    arow = arow < n ? arow : (n - 1);           // clamp; padded rows discarded
    const float* xrow = x + (size_t)arow * IN_DIM;

    v8f c[4] = {};
    #pragma unroll
    for (int ki = 0; ki < 4; ki++) {
        const int ks = ki * 32;
        const int k1 = ks + half * 16;
        const int k2 = ks + half * 16 + 8;
        float4 f0 = *(const float4*)(xrow + k1);
        float4 f1 = *(const float4*)(xrow + k1 + 4);
        float4 f2 = *(const float4*)(xrow + k2);
        float4 f3 = *(const float4*)(xrow + k2 + 4);
        v16bf a;
        a[0] = to_bf16(f0.x); a[1] = to_bf16(f0.y); a[2]  = to_bf16(f0.z); a[3]  = to_bf16(f0.w);
        a[4] = to_bf16(f1.x); a[5] = to_bf16(f1.y); a[6]  = to_bf16(f1.z); a[7]  = to_bf16(f1.w);
        a[8] = to_bf16(f2.x); a[9] = to_bf16(f2.y); a[10] = to_bf16(f2.z); a[11] = to_bf16(f2.w);
        a[12] = to_bf16(f3.x); a[13] = to_bf16(f3.y); a[14] = to_bf16(f3.z); a[15] = to_bf16(f3.w);

        // Batch all 4 N-tile B fragments (8x ds_load_b128 in one clause,
        // one dscnt wait), then 4 back-to-back WMMAs.
        const int kb = ks + half * 16;
        v16bf b0 = load_bfrag(&wlds[(0 * 16 + nloc) * S1 + kb]);
        v16bf b1 = load_bfrag(&wlds[(1 * 16 + nloc) * S1 + kb]);
        v16bf b2 = load_bfrag(&wlds[(2 * 16 + nloc) * S1 + kb]);
        v16bf b3 = load_bfrag(&wlds[(3 * 16 + nloc) * S1 + kb]);
        c[0] = wmma_bf16(a, b0, c[0]);
        c[1] = wmma_bf16(a, b1, c[1]);
        c[2] = wmma_bf16(a, b2, c[2]);
        c[3] = wmma_bf16(a, b3, c[3]);
    }

    #pragma unroll
    for (int r = 0; r < 8; r++) {
        const int row = rowBase + half * 8 + r;
        if (row < n) {
            const float di2 = dinv[row] * dinv[row];
            #pragma unroll
            for (int nt = 0; nt < 4; nt++) {
                const int col = nt * 16 + nloc;
                float v = c[nt][r];
                h1[(size_t)row * HID + col]   = v;
                agg1[(size_t)row * HID + col] = v * di2;
            }
        }
    }
}

// ------------------------------------------------------------- scatter 1 --
// agg1[dst] += h1[src] * (dinv[src]*w*dinv[dst]); 64 lanes per edge (1 row = 256B)
__global__ __launch_bounds__(256) void scatter1_kernel(
    const int* __restrict__ src, const int* __restrict__ dst,
    const float* __restrict__ w, const float* __restrict__ dinv,
    const float* __restrict__ h1, float* __restrict__ agg1, int E)
{
    int gid = blockIdx.x * 256 + threadIdx.x;
    int e = gid >> 6, f = gid & 63;
    if (e >= E) return;
    int s = src[e], d = dst[e];
    float nrm = dinv[s] * w[e] * dinv[d];
    atomicAdd(&agg1[(size_t)d * HID + f], h1[(size_t)s * HID + f] * nrm);
}

// --------------------------------------------------------------- GEMM 2 ---
// h3 = relu(agg1 + b1) @ W2 ; out = h3 * dinv^2 + b2 (self-loop seed + bias)
__global__ __launch_bounds__(256) void gemm2_kernel(
    const float* __restrict__ agg1, const float* __restrict__ b1,
    const float* __restrict__ W2, const float* __restrict__ b2,
    const float* __restrict__ dinv,
    float* __restrict__ h3, float* __restrict__ out, int n)
{
    __shared__ __bf16 wlds[EMB * S2];   // column-major W2: wlds[col*S2 + k]
    __shared__ float  b1s[HID];
    for (int i = threadIdx.x; i < HID * EMB; i += 256) {
        int k = i >> 7, col = i & 127;          // W2 is [k][col], row-major
        wlds[col * S2 + k] = to_bf16(W2[i]);
    }
    if (threadIdx.x < HID) b1s[threadIdx.x] = b1[threadIdx.x];
    __syncthreads();

    const int wave = threadIdx.x >> 5;
    const int lane = threadIdx.x & 31;
    const int half = lane >> 4;
    const int nloc = lane & 15;
    const int rowBase = blockIdx.x * 128 + wave * 16;

    int arow = rowBase + (lane & 15);
    arow = arow < n ? arow : (n - 1);
    const float* hrow = agg1 + (size_t)arow * HID;

    v8f c[8] = {};
    #pragma unroll
    for (int ki = 0; ki < 2; ki++) {
        const int ks = ki * 32;
        const int k1 = ks + half * 16;
        const int k2 = ks + half * 16 + 8;
        v16bf a;
        #pragma unroll
        for (int i = 0; i < 8; i++)
            a[i] = to_bf16(fmaxf(hrow[k1 + i] + b1s[k1 + i], 0.0f));
        #pragma unroll
        for (int i = 0; i < 8; i++)
            a[8 + i] = to_bf16(fmaxf(hrow[k2 + i] + b1s[k2 + i], 0.0f));

        const int kb = ks + half * 16;
        // Two batches of 4 fragments to cap VGPR pressure (~140 regs).
        {
            v16bf b0 = load_bfrag(&wlds[(0 * 16 + nloc) * S2 + kb]);
            v16bf b1f = load_bfrag(&wlds[(1 * 16 + nloc) * S2 + kb]);
            v16bf b2f = load_bfrag(&wlds[(2 * 16 + nloc) * S2 + kb]);
            v16bf b3f = load_bfrag(&wlds[(3 * 16 + nloc) * S2 + kb]);
            c[0] = wmma_bf16(a, b0, c[0]);
            c[1] = wmma_bf16(a, b1f, c[1]);
            c[2] = wmma_bf16(a, b2f, c[2]);
            c[3] = wmma_bf16(a, b3f, c[3]);
        }
        {
            v16bf b4 = load_bfrag(&wlds[(4 * 16 + nloc) * S2 + kb]);
            v16bf b5 = load_bfrag(&wlds[(5 * 16 + nloc) * S2 + kb]);
            v16bf b6 = load_bfrag(&wlds[(6 * 16 + nloc) * S2 + kb]);
            v16bf b7 = load_bfrag(&wlds[(7 * 16 + nloc) * S2 + kb]);
            c[4] = wmma_bf16(a, b4, c[4]);
            c[5] = wmma_bf16(a, b5, c[5]);
            c[6] = wmma_bf16(a, b6, c[6]);
            c[7] = wmma_bf16(a, b7, c[7]);
        }
    }

    #pragma unroll
    for (int r = 0; r < 8; r++) {
        const int row = rowBase + half * 8 + r;
        if (row < n) {
            const float di2 = dinv[row] * dinv[row];
            #pragma unroll
            for (int nt = 0; nt < 8; nt++) {
                const int col = nt * 16 + nloc;
                float v = c[nt][r];
                h3[(size_t)row * EMB + col]  = v;
                out[(size_t)row * EMB + col] = v * di2 + b2[col];
            }
        }
    }
}

// ------------------------------------------------------------- scatter 2 --
__global__ __launch_bounds__(256) void scatter2_kernel(
    const int* __restrict__ src, const int* __restrict__ dst,
    const float* __restrict__ w, const float* __restrict__ dinv,
    const float* __restrict__ h3, float* __restrict__ out, int E)
{
    int gid = blockIdx.x * 256 + threadIdx.x;
    int e = gid >> 7, f = gid & 127;
    if (e >= E) return;
    int s = src[e], d = dst[e];
    float nrm = dinv[s] * w[e] * dinv[d];
    atomicAdd(&out[(size_t)d * EMB + f], h3[(size_t)s * EMB + f] * nrm);
}

// ---------------------------------------------------------------- launch --
extern "C" void kernel_launch(void* const* d_in, const int* in_sizes, int n_in,
                              void* d_out, int out_size, void* d_ws, size_t ws_size,
                              hipStream_t stream)
{
    const float* x  = (const float*)d_in[0];
    const int*   ei = (const int*)d_in[1];
    const float* ew = (const float*)d_in[2];
    const float* W1 = (const float*)d_in[3];
    const float* b1 = (const float*)d_in[4];
    const float* W2 = (const float*)d_in[5];
    const float* b2 = (const float*)d_in[6];

    const int n = in_sizes[0] / IN_DIM;
    const int E = in_sizes[1] / 2;
    const int* src = ei;        // edge_index[0]
    const int* dst = ei + E;    // edge_index[1]

    float* ws   = (float*)d_ws;
    float* deg  = ws;                          // n floats (becomes dinv in place)
    float* h1   = deg  + n;                    // n*64
    float* agg1 = h1   + (size_t)n * HID;      // n*64
    float* h3   = agg1 + (size_t)n * HID;      // n*128
    float* out  = (float*)d_out;

    init_deg_kernel <<<(n + 255) / 256, 256, 0, stream>>>(deg, n);
    deg_accum_kernel<<<(E + 255) / 256, 256, 0, stream>>>(dst, ew, deg, E);
    dinv_kernel     <<<(n + 255) / 256, 256, 0, stream>>>(deg, n);

    gemm1_kernel   <<<(n + 127) / 128, 256, 0, stream>>>(x, W1, deg, h1, agg1, n);
    scatter1_kernel<<<(E + 3) / 4,     256, 0, stream>>>(src, dst, ew, deg, h1, agg1, E);
    gemm2_kernel   <<<(n + 127) / 128, 256, 0, stream>>>(agg1, b1, W2, b2, deg, h3, out, n);
    scatter2_kernel<<<(E + 1) / 2,     256, 0, stream>>>(src, dst, ew, deg, h3, out, E);
}
